// OhioModel_56478819942660
// MI455X (gfx1250) — matrix operand order repaired and checked by reading, not compiled, not run on hardware
//
#include <hip/hip_runtime.h>

// ---------------- model constants (from reference) ----------------
#define B_    64
#define T_    192
#define PRE_  160
#define DIN_  8
#define DM_   256
#define H_    8
#define DH_   32
#define DFF_  1024
#define L_    4
#define SCALE_ 0.1767766952966369f   // 1/sqrt(32)

typedef __bf16 bf16_t;
typedef __attribute__((ext_vector_type(16))) __bf16        v16bf;
typedef __attribute__((ext_vector_type(8)))  float         v8f;
typedef __attribute__((ext_vector_type(4)))  unsigned int  uint4v;

union FragBF { v16bf v; uint4v q[2]; };

__device__ __forceinline__ bf16_t f2bf(float f) {
    union { float f; unsigned int u; } a; a.f = f;
    unsigned int r = (a.u + 0x7FFFu + ((a.u >> 16) & 1u)) >> 16;   // RNE
    union { unsigned short s; __bf16 b; } o; o.s = (unsigned short)r;
    return o.b;
}
__device__ __forceinline__ float bf2f(bf16_t v) {
    union { unsigned short s; __bf16 b; } i; i.b = v;
    union { unsigned int u; float f; } o; o.u = ((unsigned int)i.s) << 16;
    return o.f;
}

// ---------------- utility kernels ----------------
__global__ void copy_f32_kernel(const float* __restrict__ src, float* __restrict__ dst, int n) {
    int i = blockIdx.x * blockDim.x + threadIdx.x;
    if (i < n) dst[i] = src[i];
}

// W [K][N] f32  ->  WT [N][K] bf16
__global__ void transpose_w_kernel(const float* __restrict__ W, bf16_t* __restrict__ WT,
                                   int K, int N) {
    int i = blockIdx.x * blockDim.x + threadIdx.x;
    if (i >= K * N) return;
    int n = i / K, k = i % K;
    WT[i] = f2bf(W[k * N + n]);
}

// bqkv[l][0:256]=bq[l], [256:512]=bk[l], [512:768]=bv[l]
__global__ void concat_bias_kernel(const float* __restrict__ bq, const float* __restrict__ bk,
                                   const float* __restrict__ bv, float* __restrict__ bqkv) {
    int i = blockIdx.x * blockDim.x + threadIdx.x;
    if (i >= L_ * 3 * DM_) return;
    int l = i / (3 * DM_);
    int r = i % (3 * DM_);
    int seg = r >> 8, n = r & 255;
    const float* src = (seg == 0) ? bq : (seg == 1) ? bk : bv;
    bqkv[i] = src[l * DM_ + n];
}

// x = d_out buffer [B][T][8]; tokens t0..t0+ntok-1; h rows = b*ntok + ti
__global__ void embed_kernel(const float* __restrict__ x, int t0, int ntok,
                             const float* __restrict__ emb_w,  // [8][256]
                             const float* __restrict__ emb_b,  // [256]
                             float* __restrict__ hf, bf16_t* __restrict__ hb) {
    int idx = blockIdx.x * blockDim.x + threadIdx.x;
    int total = B_ * ntok * DM_;
    if (idx >= total) return;
    int d  = idx % DM_;
    int r  = idx / DM_;
    int ti = r % ntok;
    int b  = r / ntok;
    const float* xin = x + ((size_t)b * T_ + (t0 + ti)) * DIN_;
    float s = emb_b[d];
    #pragma unroll
    for (int i = 0; i < DIN_; ++i) s += xin[i] * emb_w[i * DM_ + d];
    int row = b * ntok + ti;
    hf[(size_t)row * DM_ + d] = s;
    hb[(size_t)row * DM_ + d] = f2bf(s);
}

// ---------------- WMMA bf16 GEMM:  C[M,N] = A[M,K] * (WT[N][K])^T + bias ----------------
// one wave computes a 64x16 strip (4 M-tiles); B fragment reused 4x per K-step.
// qkv mode: N=768 output routed by n>>8 into (Cb:q, Cb1:Kcache, Cb2:Vcache).
__global__ void gemm_bf16_kernel(const bf16_t* __restrict__ A, int lda,
                                 const bf16_t* __restrict__ WT, int ldw,   // ldw == K
                                 const float* __restrict__ bias,
                                 int M, int N, int K,
                                 float* __restrict__ Cf, int ldcf,
                                 bf16_t* __restrict__ Cb, int ldcb,
                                 bf16_t* __restrict__ Cb1, bf16_t* __restrict__ Cb2, int ldcb12,
                                 int relu, int qkv) {
    int wave = threadIdx.x >> 5;
    int lane = threadIdx.x & 31;
    int tilesM4 = M >> 6, tilesN = N >> 4;
    int tile = blockIdx.x * 4 + wave;
    if (tile >= tilesM4 * tilesN) return;
    int tm = tile % tilesM4;          // 64-row supertile index
    int tn = tile / tilesM4;
    int r    = lane & 15;             // A row in tile / WT row in tile / D column
    int half = lane >> 4;

    const bf16_t* Ar[4];
    #pragma unroll
    for (int i = 0; i < 4; ++i)
        Ar[i] = A + (size_t)(tm * 64 + i * 16 + r) * lda;
    const bf16_t* Brow = WT + (size_t)(tn * 16 + r) * ldw;

    v8f acc[4] = {};
    for (int kk = 0; kk < K; kk += 32) {
        FragBF b;
        // B 32x16: lane holds column n=r, rows k = half*8.. and 16+half*8.. (same as A)
        b.q[0] = *(const uint4v*)(Brow + kk + half * 8);
        b.q[1] = *(const uint4v*)(Brow + kk + 16 + half * 8);
        #pragma unroll
        for (int i = 0; i < 4; ++i) {
            FragBF a;
            // 16-bit A 16x32 layout: lane holds K = half*8.. and 16+half*8..
            a.q[0] = *(const uint4v*)(Ar[i] + kk + half * 8);
            a.q[1] = *(const uint4v*)(Ar[i] + kk + 16 + half * 8);
            acc[i] = __builtin_amdgcn_wmma_f32_16x16x32_bf16(false, a.v, false, b.v,
                                                             (short)0, acc[i], false, false);
        }
    }

    int n = tn * 16 + r;                      // D: lane -> column n = lane%16
    float bv = bias ? bias[n] : 0.0f;
    if (qkv) {
        int seg = n >> 8, nn = n & 255;
        bf16_t* dst = (seg == 0) ? Cb : (seg == 1) ? Cb1 : Cb2;
        int ld = (seg == 0) ? ldcb : ldcb12;
        #pragma unroll
        for (int i = 0; i < 4; ++i) {
            #pragma unroll
            for (int g = 0; g < 8; ++g) {     // D: VGPR g -> row g + half*8
                int m = tm * 64 + i * 16 + g + half * 8;
                dst[(size_t)m * ld + nn] = f2bf(acc[i][g] + bv);
            }
        }
    } else {
        #pragma unroll
        for (int i = 0; i < 4; ++i) {
            #pragma unroll
            for (int g = 0; g < 8; ++g) {
                int m = tm * 64 + i * 16 + g + half * 8;
                float v = acc[i][g] + bv;
                if (relu) v = fmaxf(v, 0.0f);
                if (Cf) Cf[(size_t)m * ldcf + n] = v;
                if (Cb) Cb[(size_t)m * ldcb + n] = f2bf(v);
            }
        }
    }
}

// ---------------- attention: one wave per (b, head, query); lane = head dim ----------------
__global__ void attn_kernel(const bf16_t* __restrict__ qb, int nq, int qstride, int key_off,
                            const bf16_t* __restrict__ kc,   // layer base [B][T][DM]
                            const bf16_t* __restrict__ vc,
                            bf16_t* __restrict__ outb, int nwaves) {
    int w = blockIdx.x * (blockDim.x >> 5) + (threadIdx.x >> 5);
    if (w >= nwaves) return;
    int lane = threadIdx.x & 31;
    int t  = w % nq;
    int hh = (w / nq) % H_;
    int b  = w / (nq * H_);
    int pos  = key_off + t;
    int qrow = b * qstride + t;

    float q = bf2f(qb[(size_t)qrow * DM_ + hh * DH_ + lane]) * SCALE_;
    const bf16_t* kbase = kc + (size_t)b * T_ * DM_ + hh * DH_ + lane;
    const bf16_t* vbase = vc + (size_t)b * T_ * DM_ + hh * DH_ + lane;

    float m = -3.0e38f, lsum = 0.0f, acc = 0.0f;
    for (int tk = 0; tk <= pos; ++tk) {
        float s = q * bf2f(kbase[(size_t)tk * DM_]);
        #pragma unroll
        for (int d = 16; d >= 1; d >>= 1) s += __shfl_xor(s, d, 32);
        float mn = fmaxf(m, s);
        float c  = __expf(m - mn);
        float p  = __expf(s - mn);
        lsum = lsum * c + p;
        acc  = acc  * c + p * bf2f(vbase[(size_t)tk * DM_]);
        m = mn;
    }
    outb[(size_t)qrow * DM_ + hh * DH_ + lane] = f2bf(acc / lsum);
}

// ---------------- fused residual + LayerNorm; one wave per row ----------------
__global__ void ln_kernel(float* __restrict__ hf, const float* __restrict__ proj,
                          const float* __restrict__ g, const float* __restrict__ beta,
                          int M, bf16_t* __restrict__ hb) {
    int row = blockIdx.x * (blockDim.x >> 5) + (threadIdx.x >> 5);
    if (row >= M) return;
    int lane = threadIdx.x & 31;
    float x[8]; float s = 0.0f;
    #pragma unroll
    for (int j = 0; j < 8; ++j) {
        int d = lane + j * 32;
        x[j] = hf[(size_t)row * DM_ + d] + proj[(size_t)row * DM_ + d];
        s += x[j];
    }
    #pragma unroll
    for (int d = 16; d >= 1; d >>= 1) s += __shfl_xor(s, d, 32);
    float mean = s * (1.0f / DM_);
    float vs = 0.0f;
    #pragma unroll
    for (int j = 0; j < 8; ++j) { float t = x[j] - mean; vs += t * t; }
    #pragma unroll
    for (int d = 16; d >= 1; d >>= 1) vs += __shfl_xor(vs, d, 32);
    float rstd = rsqrtf(vs * (1.0f / DM_) + 1e-5f);
    #pragma unroll
    for (int j = 0; j < 8; ++j) {
        int d = lane + j * 32;
        float y = (x[j] - mean) * rstd * g[d] + beta[d];
        hf[(size_t)row * DM_ + d] = y;
        hb[(size_t)row * DM_ + d] = f2bf(y);
    }
}

// ---------------- final projection: pred = h . fin_w + fin_b -> out[b, wpos, 0] ----------------
__global__ void final_kernel(const float* __restrict__ hf, int rowstride, int rowoff,
                             const float* __restrict__ fin_w, const float* __restrict__ fin_b,
                             float* __restrict__ out, int wpos) {
    int b = blockIdx.x, lane = threadIdx.x;
    int row = b * rowstride + rowoff;
    float s = 0.0f;
    #pragma unroll
    for (int j = 0; j < 8; ++j) {
        int d = lane + j * 32;
        s += hf[(size_t)row * DM_ + d] * fin_w[d];
    }
    #pragma unroll
    for (int d = 16; d >= 1; d >>= 1) s += __shfl_xor(s, d, 32);
    if (lane == 0) out[((size_t)b * T_ + wpos) * DIN_] = s + fin_b[0];
}

// ---------------- host orchestration ----------------
extern "C" void kernel_launch(void* const* d_in, const int* in_sizes, int n_in,
                              void* d_out, int out_size, void* d_ws, size_t ws_size,
                              hipStream_t stream) {
    (void)in_sizes; (void)n_in; (void)out_size; (void)ws_size;

    const float* x_in  = (const float*)d_in[0];
    const float* emb_w = (const float*)d_in[1];
    const float* emb_b = (const float*)d_in[2];
    const float* Wq = (const float*)d_in[3];  const float* bq = (const float*)d_in[4];
    const float* Wk = (const float*)d_in[5];  const float* bk = (const float*)d_in[6];
    const float* Wv = (const float*)d_in[7];  const float* bv = (const float*)d_in[8];
    const float* Wo = (const float*)d_in[9];  const float* bo = (const float*)d_in[10];
    const float* ln1_g = (const float*)d_in[11]; const float* ln1_b = (const float*)d_in[12];
    const float* W1 = (const float*)d_in[13]; const float* b1 = (const float*)d_in[14];
    const float* W2 = (const float*)d_in[15]; const float* b2 = (const float*)d_in[16];
    const float* ln2_g = (const float*)d_in[17]; const float* ln2_b = (const float*)d_in[18];
    const float* fin_w = (const float*)d_in[19]; const float* fin_b = (const float*)d_in[20];
    float* out = (float*)d_out;

    // ---- carve workspace ----
    char* ws = (char*)d_ws; size_t off = 0;
    auto carve = [&](size_t bytes) -> void* {
        void* p = ws + off;
        off += (bytes + 255) & ~(size_t)255;
        return p;
    };
    const size_t BTD = (size_t)B_ * T_ * DM_;
    bf16_t* wqkvT = (bf16_t*)carve((size_t)L_ * 3 * DM_ * DM_ * 2);  // [L][3*256][256]
    bf16_t* woT   = (bf16_t*)carve((size_t)L_ * DM_ * DM_ * 2);
    bf16_t* w1T   = (bf16_t*)carve((size_t)L_ * DM_ * DFF_ * 2);
    bf16_t* w2T   = (bf16_t*)carve((size_t)L_ * DFF_ * DM_ * 2);
    float*  bqkv  = (float*) carve((size_t)L_ * 3 * DM_ * 4);
    bf16_t* kc    = (bf16_t*)carve((size_t)L_ * BTD * 2);
    bf16_t* vc    = (bf16_t*)carve((size_t)L_ * BTD * 2);
    float*  hf    = (float*) carve(BTD * 4);
    bf16_t* hb    = (bf16_t*)carve(BTD * 2);
    bf16_t* qb    = (bf16_t*)carve(BTD * 2);
    bf16_t* attnb = (bf16_t*)carve(BTD * 2);
    float*  proj  = (float*) carve(BTD * 4);
    bf16_t* ffb   = (bf16_t*)carve((size_t)B_ * T_ * DFF_ * 2);

    auto xpose = [&](const float* W, bf16_t* WT, int K, int N) {
        int n = K * N;
        transpose_w_kernel<<<(n + 255) / 256, 256, 0, stream>>>(W, WT, K, N);
    };
    auto gemm = [&](const bf16_t* A, int lda, const bf16_t* WT, const float* bias,
                    int M, int N, int K, float* Cf, int ldcf, bf16_t* Cb, int ldcb,
                    bf16_t* Cb1, bf16_t* Cb2, int ldcb12, int relu, int qkv) {
        int tiles = (M / 64) * (N / 16);
        gemm_bf16_kernel<<<(tiles + 3) / 4, 128, 0, stream>>>(
            A, lda, WT, K, bias, M, N, K, Cf, ldcf, Cb, ldcb, Cb1, Cb2, ldcb12, relu, qkv);
    };

    // ---- weight prep (every call; deterministic) ----
    for (int l = 0; l < L_; ++l) {
        bf16_t* wl = wqkvT + (size_t)l * 3 * DM_ * DM_;
        xpose(Wq + (size_t)l * DM_ * DM_, wl,                 DM_, DM_);
        xpose(Wk + (size_t)l * DM_ * DM_, wl + DM_ * DM_,     DM_, DM_);
        xpose(Wv + (size_t)l * DM_ * DM_, wl + 2 * DM_ * DM_, DM_, DM_);
        xpose(Wo + (size_t)l * DM_ * DM_, woT + (size_t)l * DM_ * DM_, DM_, DM_);
        xpose(W1 + (size_t)l * DM_ * DFF_, w1T + (size_t)l * DFF_ * DM_, DM_, DFF_);
        xpose(W2 + (size_t)l * DFF_ * DM_, w2T + (size_t)l * DM_ * DFF_, DFF_, DM_);
    }
    concat_bias_kernel<<<(L_ * 3 * DM_ + 255) / 256, 256, 0, stream>>>(bq, bk, bv, bqkv);

    // ---- copy input buffer to output (generated tokens appended in-place there) ----
    {
        int n = B_ * T_ * DIN_;
        copy_f32_kernel<<<(n + 255) / 256, 256, 0, stream>>>(x_in, out, n);
    }

    // ================= PREFILL (tokens 0..T-1; only <PRE matter causally) =================
    {
        int M = B_ * T_;
        embed_kernel<<<(M * DM_ + 255) / 256, 256, 0, stream>>>(out, 0, T_, emb_w, emb_b, hf, hb);
        for (int l = 0; l < L_; ++l) {
            const bf16_t* wqkv_l = wqkvT + (size_t)l * 3 * DM_ * DM_;
            const bf16_t* wo_l   = woT + (size_t)l * DM_ * DM_;
            bf16_t* kc_l = kc + (size_t)l * BTD;
            bf16_t* vc_l = vc + (size_t)l * BTD;
            // fused QKV: N=768, q->qb, k->Kcache, v->Vcache
            gemm(hb, DM_, wqkv_l, bqkv + l * 3 * DM_, M, 3 * DM_, DM_,
                 nullptr, 0, qb, DM_, kc_l, vc_l, DM_, 0, 1);
            int nw = B_ * H_ * PRE_;
            attn_kernel<<<(nw + 7) / 8, 256, 0, stream>>>(qb, PRE_, T_, 0, kc_l, vc_l, attnb, nw);
            gemm(attnb, DM_, wo_l, bo + l * DM_, M, DM_, DM_, proj, DM_, nullptr, 0,
                 nullptr, nullptr, 0, 0, 0);
            ln_kernel<<<(M + 7) / 8, 256, 0, stream>>>(hf, proj, ln1_g + l * DM_, ln1_b + l * DM_, M, hb);
            gemm(hb,  DM_,  w1T + (size_t)l * DFF_ * DM_, b1 + l * DFF_, M, DFF_, DM_,
                 nullptr, 0, ffb, DFF_, nullptr, nullptr, 0, 1, 0);
            gemm(ffb, DFF_, w2T + (size_t)l * DM_ * DFF_, b2 + l * DM_,  M, DM_, DFF_,
                 proj, DM_, nullptr, 0, nullptr, nullptr, 0, 0, 0);
            ln_kernel<<<(M + 7) / 8, 256, 0, stream>>>(hf, proj, ln2_g + l * DM_, ln2_b + l * DM_, M, hb);
        }
        // prediction from token PRE-1 -> buf[:, PRE, 0]
        final_kernel<<<B_, 32, 0, stream>>>(hf, T_, PRE_ - 1, fin_w, fin_b, out, PRE_);
    }

    // ================= DECODE: 31 steps, token at p = 160..190 =================
    for (int p = PRE_; p < T_ - 1; ++p) {
        int M = B_;
        embed_kernel<<<(M * DM_ + 255) / 256, 256, 0, stream>>>(out, p, 1, emb_w, emb_b, hf, hb);
        for (int l = 0; l < L_; ++l) {
            const bf16_t* wqkv_l = wqkvT + (size_t)l * 3 * DM_ * DM_;
            const bf16_t* wo_l   = woT + (size_t)l * DM_ * DM_;
            bf16_t* kc_l = kc + (size_t)l * BTD;
            bf16_t* vc_l = vc + (size_t)l * BTD;
            // fused QKV; K/V appended straight into the cache at position p
            gemm(hb, DM_, wqkv_l, bqkv + l * 3 * DM_, M, 3 * DM_, DM_,
                 nullptr, 0, qb, DM_, kc_l + (size_t)p * DM_, vc_l + (size_t)p * DM_, T_ * DM_, 0, 1);
            int nw = B_ * H_;
            attn_kernel<<<(nw + 7) / 8, 256, 0, stream>>>(qb, 1, 1, p, kc_l, vc_l, attnb, nw);
            gemm(attnb, DM_, wo_l, bo + l * DM_, M, DM_, DM_, proj, DM_, nullptr, 0,
                 nullptr, nullptr, 0, 0, 0);
            ln_kernel<<<(M + 7) / 8, 256, 0, stream>>>(hf, proj, ln1_g + l * DM_, ln1_b + l * DM_, M, hb);
            gemm(hb,  DM_,  w1T + (size_t)l * DFF_ * DM_, b1 + l * DFF_, M, DFF_, DM_,
                 nullptr, 0, ffb, DFF_, nullptr, nullptr, 0, 1, 0);
            gemm(ffb, DFF_, w2T + (size_t)l * DM_ * DFF_, b2 + l * DM_,  M, DM_, DFF_,
                 proj, DM_, nullptr, 0, nullptr, nullptr, 0, 0, 0);
            ln_kernel<<<(M + 7) / 8, 256, 0, stream>>>(hf, proj, ln2_g + l * DM_, ln2_b + l * DM_, M, hb);
        }
        final_kernel<<<B_, 32, 0, stream>>>(hf, 1, 0, fin_w, fin_b, out, p + 1);
    }
}
